// MoeLayer_72112500900499
// MI455X (gfx1250) — compile-verified, matrix operand order checked by reading, not run on hardware
//
#include <hip/hip_runtime.h>
#include <hip/hip_bf16.h>
#include <math.h>

#define T_TOK 1024
#define D_DIM 2048
#define H_DIM 7168
#define E_NUM 8
#define K_TOP 2

#define MACRO_M 64           // rows per macro tile (one expert each, segments 64-aligned)
#define SUB_M   16
#define NSUB    (MACRO_M / SUB_M)
#define CAP     2560         // 2048 pairs + up to 8*64 padding, 64-aligned
#define MT      (CAP / MACRO_M)   // 40 macro M-tiles
#define BN      64           // N columns per block (4 waves x 16)
#define KSTEP   32           // bf16 WMMA K

typedef __attribute__((ext_vector_type(16))) __bf16 v16bf;
typedef __attribute__((ext_vector_type(8)))  __bf16 v8bf;
typedef __attribute__((ext_vector_type(8)))  float  v8f;

// ---------------- routing kernels ----------------

__global__ void k_init(float* out, int* counts, int* pair_tok, float* pair_w) {
    int i = blockIdx.x * blockDim.x + threadIdx.x;
    if (i < T_TOK * D_DIM) out[i] = 0.0f;
    if (i < E_NUM) counts[i] = 0;
    if (i < CAP) { pair_tok[i] = -1; pair_w[i] = 0.0f; }
}

__global__ void k_count(const int* __restrict__ sel, int* counts) {
    int i = blockIdx.x * blockDim.x + threadIdx.x;
    if (i < T_TOK * K_TOP) atomicAdd(&counts[sel[i]], 1);
}

__global__ void k_scan(const int* __restrict__ counts, int* offsets, int* cursor) {
    if (blockIdx.x == 0 && threadIdx.x == 0) {
        int off = 0;
        for (int e = 0; e < E_NUM; ++e) {
            offsets[e] = off;
            cursor[e]  = off;
            off += ((counts[e] + MACRO_M - 1) / MACRO_M) * MACRO_M; // 64-align segments
        }
        offsets[E_NUM] = off;
    }
}

__global__ void k_scatter(const int* __restrict__ sel, const float* __restrict__ w,
                          int* cursor, int* pair_tok, float* pair_w) {
    int i = blockIdx.x * blockDim.x + threadIdx.x;
    if (i < T_TOK * K_TOP) {
        int e   = sel[i];
        int pos = atomicAdd(&cursor[e], 1);
        pair_tok[pos] = i / K_TOP;
        pair_w[pos]   = w[i];
    }
}

// Build a 16-wide bf16 fragment from two aligned 8-wide LDS vectors.
__device__ inline v16bf frag_from_lds(const __bf16* base) {
    v8bf lo = *(const v8bf*)(base);
    v8bf hi = *(const v8bf*)(base + 16);
    return __builtin_shufflevector(lo, hi, 0, 1, 2, 3, 4, 5, 6, 7,
                                   8, 9, 10, 11, 12, 13, 14, 15);
}

// ---------------- pass 1: h = silu(x W1^T) * (x W3^T), written bf16 ----------------

__global__ __launch_bounds__(128) void k_gemm13(
    const float* __restrict__ x, const float* __restrict__ w1, const float* __restrict__ w3,
    const int* __restrict__ offsets, const int* __restrict__ pair_tok,
    __bf16* __restrict__ hbuf)
{
    __shared__ __bf16 tileA[MACRO_M * KSTEP];      // 64 rows x 32 k, bf16, 4 KB

    const int m0    = blockIdx.y * MACRO_M;
    const int total = offsets[E_NUM];
    if (m0 >= total) return;                       // uniform per block
    int e = 0;
    while (e < E_NUM - 1 && m0 >= offsets[e + 1]) ++e;

    const int tid  = threadIdx.x;
    const int lane = tid & 31;
    const int wave = tid >> 5;
    const int nrow = lane & 15;                    // N (hh) within wave tile
    const int half = lane >> 4;                    // K-half select
    const int kb0  = half * 8;
    const int n0   = blockIdx.x * BN + wave * 16;  // hh base for this wave

    const size_t wslice = (size_t)e * H_DIM * D_DIM;
    const float* b1row  = w1 + wslice + (size_t)(n0 + nrow) * D_DIM;
    const float* b3row  = w3 + wslice + (size_t)(n0 + nrow) * D_DIM;

    // Cooperative A staging: thread -> (row = tid>>1, 16-col half = (tid&1)*16).
    // Padded rows (tok < 0) just load token 0: their h output is discarded in
    // pass 2 (tok>=0 guard there), so no branch is needed in this hot loop.
    const int fill_row  = tid >> 1;
    const int fill_col  = (tid & 1) * 16;
    int ftok = pair_tok[m0 + fill_row];
    if (ftok < 0) ftok = 0;
    const float* frow = x + (size_t)ftok * D_DIM + fill_col;
    __bf16* fdst = &tileA[fill_row * KSTEP + fill_col];

    v8f acc1[NSUB] = {};
    v8f acc3[NSUB] = {};

    for (int k0 = 0; k0 < D_DIM; k0 += KSTEP) {
        // ---- stage A tile (64x32 fp32 -> bf16 in LDS) ----
        __syncthreads();                           // previous iter's frags consumed
        {
            const float4* p = (const float4*)(frow + k0);
            float4 q0 = p[0], q1 = p[1], q2 = p[2], q3 = p[3];
            v8bf lo, hi;
            lo[0] = (__bf16)q0.x; lo[1] = (__bf16)q0.y; lo[2] = (__bf16)q0.z; lo[3] = (__bf16)q0.w;
            lo[4] = (__bf16)q1.x; lo[5] = (__bf16)q1.y; lo[6] = (__bf16)q1.z; lo[7] = (__bf16)q1.w;
            hi[0] = (__bf16)q2.x; hi[1] = (__bf16)q2.y; hi[2] = (__bf16)q2.z; hi[3] = (__bf16)q2.w;
            hi[4] = (__bf16)q3.x; hi[5] = (__bf16)q3.y; hi[6] = (__bf16)q3.z; hi[7] = (__bf16)q3.w;
            *(v8bf*)(fdst)     = lo;
            *(v8bf*)(fdst + 8) = hi;
        }
        __syncthreads();

        // ---- load B fragments (w1/w3 rows, fp32 -> bf16) ----
        const int kb = k0 + kb0;
        __builtin_prefetch(b1row + kb + 2 * KSTEP, 0, 1);   // global_prefetch_b8
        __builtin_prefetch(b3row + kb + 2 * KSTEP, 0, 1);
        v16bf b1, b3;
#pragma unroll
        for (int i = 0; i < 8; ++i) {
            b1[i]     = (__bf16)b1row[kb + i];
            b1[i + 8] = (__bf16)b1row[kb + 16 + i];
            b3[i]     = (__bf16)b3row[kb + i];
            b3[i + 8] = (__bf16)b3row[kb + 16 + i];
        }

        // ---- 4 M-subtiles reuse the B fragments ----
#pragma unroll
        for (int s = 0; s < NSUB; ++s) {
            v16bf a = frag_from_lds(&tileA[(s * SUB_M + nrow) * KSTEP + kb0]);
            acc1[s] = __builtin_amdgcn_wmma_f32_16x16x32_bf16(false, a, false, b1,
                                                              (short)0, acc1[s], false, false);
            acc3[s] = __builtin_amdgcn_wmma_f32_16x16x32_bf16(false, a, false, b3,
                                                              (short)0, acc3[s], false, false);
        }
    }

    // SwiGLU epilogue: D-matrix layout lane N = lane&15, M = v + 8*half.
    // silu via v_rcp_f32 (activation precision is ample; avoids IEEE div expansion).
#pragma unroll
    for (int s = 0; s < NSUB; ++s) {
#pragma unroll
        for (int v = 0; v < 8; ++v) {
            int   r   = m0 + s * SUB_M + v + half * 8;
            float g   = acc1[s][v];
            float sig = __builtin_amdgcn_rcpf(1.0f + __expf(-g));
            float h   = g * sig * acc3[s][v];
            hbuf[(size_t)r * H_DIM + n0 + nrow] = (__bf16)h;
        }
    }
}

// ---------------- pass 2: y = h W2^T, scaled scatter-add to out ----------------
// A tile (already bf16) is DMA'd global->LDS with async loads, double-buffered
// so the copy of tile k+1 overlaps the WMMAs of tile k.

__device__ inline void async_fill_tile(unsigned lds_off, const __bf16* gsrc) {
    // Per-lane 32B: INST_OFFSET is added to both the LDS and global addresses.
    asm volatile("global_load_async_to_lds_b128 %0, %1, off\n\t"
                 "global_load_async_to_lds_b128 %0, %1, off offset:16"
                 :: "v"(lds_off), "v"(gsrc) : "memory");
}

__global__ __launch_bounds__(128) void k_gemm2(
    const __bf16* __restrict__ hbuf, const float* __restrict__ w2,
    const int* __restrict__ offsets, const int* __restrict__ pair_tok,
    const float* __restrict__ pair_w, float* __restrict__ out)
{
    __shared__ __bf16 tileA[2][MACRO_M * KSTEP];   // double-buffered, 2 x 4 KB

    const int m0    = blockIdx.y * MACRO_M;
    const int total = offsets[E_NUM];
    if (m0 >= total) return;
    int e = 0;
    while (e < E_NUM - 1 && m0 >= offsets[e + 1]) ++e;

    const int tid  = threadIdx.x;
    const int lane = tid & 31;
    const int wave = tid >> 5;
    const int nrow = lane & 15;                    // N (d) within wave tile
    const int half = lane >> 4;
    const int kb0  = half * 8;
    const int n0   = blockIdx.x * BN + wave * 16;  // d base for this wave

    const float* bcol = w2 + (size_t)e * H_DIM * D_DIM + (n0 + nrow);

    // Per-thread async-copy slot: (row = tid>>1, 16-col half = (tid&1)*16) = 32B.
    const int fill_row = tid >> 1;
    const int fill_col = (tid & 1) * 16;
    const __bf16* fsrc = hbuf + (size_t)(m0 + fill_row) * H_DIM + fill_col;
    unsigned fdst_off[2];
    fdst_off[0] = (unsigned)(uintptr_t)&tileA[0][fill_row * KSTEP + fill_col];
    fdst_off[1] = (unsigned)(uintptr_t)&tileA[1][fill_row * KSTEP + fill_col];

    v8f acc[NSUB] = {};

    int buf = 0;
    async_fill_tile(fdst_off[0], fsrc);            // prime the pipeline

    for (int k0 = 0; k0 < H_DIM; k0 += KSTEP) {
        asm volatile("s_wait_asynccnt 0x0" ::: "memory");   // my fills done
        __syncthreads();                                    // everyone's fills done
        if (k0 + KSTEP < H_DIM)
            async_fill_tile(fdst_off[buf ^ 1], fsrc + k0 + KSTEP);  // overlap with WMMA

        const int kb = k0 + kb0;
        __builtin_prefetch(bcol + (size_t)(kb + 2 * KSTEP) * D_DIM, 0, 1);
        v16bf b;                                   // B[k=hh][n=d] = w2[e][hh][d] (stride-D)
#pragma unroll
        for (int i = 0; i < 8; ++i) {
            b[i]     = (__bf16)bcol[(size_t)(kb + i) * D_DIM];
            b[i + 8] = (__bf16)bcol[(size_t)(kb + 16 + i) * D_DIM];
        }
#pragma unroll
        for (int s = 0; s < NSUB; ++s) {
            v16bf a = frag_from_lds(&tileA[buf][(s * SUB_M + nrow) * KSTEP + kb0]);
            acc[s] = __builtin_amdgcn_wmma_f32_16x16x32_bf16(false, a, false, b,
                                                             (short)0, acc[s], false, false);
        }
        __syncthreads();                           // all reads of tileA[buf] done
        buf ^= 1;
    }

#pragma unroll
    for (int s = 0; s < NSUB; ++s) {
#pragma unroll
        for (int v = 0; v < 8; ++v) {
            int r   = m0 + s * SUB_M + v + half * 8;
            int tok = pair_tok[r];
            if (tok >= 0) {
                float wgt = pair_w[r];
                atomicAdd(out + (size_t)tok * D_DIM + n0 + nrow, wgt * acc[s][v]);
            }
        }
    }
}

// ---------------- launch ----------------

extern "C" void kernel_launch(void* const* d_in, const int* in_sizes, int n_in,
                              void* d_out, int out_size, void* d_ws, size_t ws_size,
                              hipStream_t stream) {
    const float* x   = (const float*)d_in[0];  // [T, D]
    const float* rw  = (const float*)d_in[1];  // [T, K]
    const int*   sel = (const int*)d_in[2];    // [T, K]
    const float* w1  = (const float*)d_in[3];  // [E, H, D]
    const float* w2  = (const float*)d_in[4];  // [E, H, D]
    const float* w3  = (const float*)d_in[5];  // [E, H, D]
    float* out = (float*)d_out;                // [T, D]

    char* ws = (char*)d_ws;
    int*    counts   = (int*)(ws + 0);
    int*    offsets  = (int*)(ws + 64);
    int*    cursor   = (int*)(ws + 128);
    int*    pair_tok = (int*)(ws + 192);
    float*  pair_w   = (float*)(ws + 192 + CAP * 4);
    __bf16* hbuf     = (__bf16*)(ws + 20736);  // CAP * H_DIM bf16 = ~36.7 MB

    k_init   <<<(T_TOK * D_DIM + 255) / 256, 256, 0, stream>>>(out, counts, pair_tok, pair_w);
    k_count  <<<(T_TOK * K_TOP + 255) / 256, 256, 0, stream>>>(sel, counts);
    k_scan   <<<1, 1, 0, stream>>>(counts, offsets, cursor);
    k_scatter<<<(T_TOK * K_TOP + 255) / 256, 256, 0, stream>>>(sel, rw, cursor, pair_tok, pair_w);

    dim3 g1(H_DIM / BN, MT);   // (112, 40)
    k_gemm13<<<g1, 128, 0, stream>>>(x, w1, w3, offsets, pair_tok, hbuf);

    dim3 g2(D_DIM / BN, MT);   // (32, 40)
    k_gemm2 <<<g2, 128, 0, stream>>>(hbuf, w2, offsets, pair_tok, pair_w, out);
}
